// GNN_2826088481036
// MI455X (gfx1250) — matrix operand order repaired and checked by reading, not compiled, not run on hardware
//
#include <hip/hip_runtime.h>

typedef _Float16 v16h __attribute__((ext_vector_type(16)));
typedef _Float16 v8h  __attribute__((ext_vector_type(8)));
typedef _Float16 h2   __attribute__((ext_vector_type(2)));
typedef float    v8f  __attribute__((ext_vector_type(8)));

__device__ __forceinline__ float softplusf(float x) {
    // stable softplus: max(x,0) + ln2 * log2(1 + exp2(-|x| * log2e))
    // raw v_exp_f32 / v_log_f32: log argument is in [1,2] (never denormal),
    // exp argument is <= 0 (never overflows), so no guard paths needed.
    float e = __builtin_amdgcn_exp2f(-fabsf(x) * 1.44269504088896f);
    float l = __builtin_amdgcn_logf(1.0f + e) * 0.69314718055995f;
    return fmaxf(x, 0.0f) + l;
}

__device__ __forceinline__ v8f wmma_f16(v16h a, v16h b, v8f c) {
    // D = A(16x32 f16) * B(32x16 f16) + C(16x16 f32)
    return __builtin_amdgcn_wmma_f32_16x16x32_f16(
        /*neg_a=*/false, a, /*neg_b=*/false, b,
        /*c_mod=*/(short)0, c, /*reuse_a=*/false, /*reuse_b=*/false);
}

__global__ __launch_bounds__(256) void gnn_collapsed_wmma(
    const float* __restrict__ x,
    const float* __restrict__ W1, const float* __restrict__ b1,
    const float* __restrict__ W2, const float* __restrict__ b2,
    const float* __restrict__ W3, const float* __restrict__ b3,
    const float* __restrict__ W4, const float* __restrict__ b4,
    const float* __restrict__ linW, const float* __restrict__ linb,
    float* __restrict__ out, int batchB, int ntiles)
{
    // per-wave private activation staging buffer: 16 rows x 32 f16 feats
    __shared__ _Float16 hbuf[8][16][32];

    const int lane = threadIdx.x & 31;
    const int wid  = threadIdx.x >> 5;
    const int row  = lane & 15;   // A-operand row (batch within tile)
    const int n    = lane & 15;   // B/D column index
    const int hi   = lane >> 4;   // upper half-wave flag

    _Float16 (*hb)[32] = hbuf[wid];

    // ---- preload weight B-operand tiles (once per wave) ----
    // Column permutation: tile t, col n holds TRUE output feature f = 2n + t,
    // so per lane (c0[r], c1[r]) are adjacent true features (2n, 2n+1).
    // B layout (32x16 f16): lanes 0-15 hold K=0..15, lanes 16-31 K=16..31,
    // halves in K order => b[e] = W[e + 16*hi][2n + t]
    v16h bw[8]; // [layer 0..3][tile 0..1]
#pragma unroll
    for (int t = 0; t < 2; ++t) {
        v16h b;
#pragma unroll
        for (int e = 0; e < 16; ++e) {
            int K = e + hi * 16;
            b[e] = (K < 4) ? (_Float16)W1[K * 32 + 2 * n + t] : (_Float16)0.0f;
        }
        bw[t] = b;
    }
    const float* Ws[3] = {W2, W3, W4};
#pragma unroll
    for (int l = 0; l < 3; ++l) {
#pragma unroll
        for (int t = 0; t < 2; ++t) {
            v16h b;
#pragma unroll
            for (int e = 0; e < 16; ++e)
                b[e] = (_Float16)Ws[l][(e + 16 * hi) * 32 + 2 * n + t];
            bw[(l + 1) * 2 + t] = b;
        }
    }

    // biases, permuted to match tile columns (feature 2n+t)
    const float bias[4][2] = {
        {b1[2 * n], b1[2 * n + 1]}, {b2[2 * n], b2[2 * n + 1]},
        {b3[2 * n], b3[2 * n + 1]}, {b4[2 * n], b4[2 * n + 1]}};
    const float sumw = linW[0] + linW[1] + linW[2];
    const float lb   = linb[0];

    const v8f zero = {};
    const int nwaves = (gridDim.x * blockDim.x) >> 5;
    const int gw = (blockIdx.x * blockDim.x + threadIdx.x) >> 5;

    for (int tile = gw; tile < ntiles; tile += nwaves) {
        // ---- build A1: per-sample node-mean of x, K padded 4 -> 32 ----
        // A layout (16x32 f16): lanes 0-15 halves 0..3 hold K=0..3; rest 0
        v16h a = {};
        const int bidx = tile * 16 + row;
        if (hi == 0 && bidx < batchB) {
            const float4* xp = reinterpret_cast<const float4*>(x) + (size_t)bidx * 3;
            float4 p0 = xp[0], p1 = xp[1], p2 = xp[2];
            const float third = 1.0f / 3.0f;
            a[0] = (_Float16)((p0.x + p1.x + p2.x) * third);
            a[1] = (_Float16)((p0.y + p1.y + p2.y) * third);
            a[2] = (_Float16)((p0.z + p1.z + p2.z) * third);
            a[3] = (_Float16)((p0.w + p1.w + p2.w) * third);
        }

        v8f c0 = zero, c1 = zero;
#pragma unroll
        for (int l = 0; l < 4; ++l) {
            c0 = wmma_f16(a, bw[l * 2 + 0], zero);   // true feats 2n
            c1 = wmma_f16(a, bw[l * 2 + 1], zero);   // true feats 2n+1
            if (l < 3) {
                // softplus, pack adjacent feature pair, single b32 DS store
#pragma unroll
                for (int r = 0; r < 8; ++r) {
                    int m = r + hi * 8;               // D row = batch row
                    h2 p;
                    p.x = (_Float16)softplusf(c0[r] + bias[l][0]);
                    p.y = (_Float16)softplusf(c1[r] + bias[l][1]);
                    *reinterpret_cast<h2*>(&hb[m][2 * n]) = p;
                }
                // A repack: K pairs are contiguous 16B runs -> two b128 DS loads
                v8h alo = *reinterpret_cast<const v8h*>(&hb[row][hi * 16]);      // K = hi*16 .. +7
                v8h ahi = *reinterpret_cast<const v8h*>(&hb[row][hi * 16 + 8]);  // K = hi*16+8 .. +15
                a = __builtin_shufflevector(alo, ahi, 0, 1, 2, 3, 4, 5, 6, 7,
                                            8, 9, 10, 11, 12, 13, 14, 15);
            }
        }

        // ---- epilogue: softplus, fold linear head, contiguous float2 stores ----
#pragma unroll
        for (int r = 0; r < 8; ++r) {
            int m = r + hi * 8;
            int b = tile * 16 + m;
            if (b < batchB) {
                float2 v;
                v.x = softplusf(c0[r] + bias[3][0]) * sumw + lb;  // feat 2n
                v.y = softplusf(c1[r] + bias[3][1]) * sumw + lb;  // feat 2n+1
                *reinterpret_cast<float2*>(out + (size_t)b * 32 + 2 * n) = v;
            }
        }
    }
}

extern "C" void kernel_launch(void* const* d_in, const int* in_sizes, int n_in,
                              void* d_out, int out_size, void* d_ws, size_t ws_size,
                              hipStream_t stream) {
    const float* x    = (const float*)d_in[0];
    const float* W1   = (const float*)d_in[1];
    const float* b1   = (const float*)d_in[2];
    const float* W2   = (const float*)d_in[3];
    const float* b2   = (const float*)d_in[4];
    const float* W3   = (const float*)d_in[5];
    const float* b3   = (const float*)d_in[6];
    const float* W4   = (const float*)d_in[7];
    const float* b4   = (const float*)d_in[8];
    const float* linW = (const float*)d_in[9];
    const float* linb = (const float*)d_in[10];
    float* out = (float*)d_out;

    const int B = in_sizes[0] / 12;          // x is [B,3,4]
    const int ntiles = (B + 15) / 16;        // 16 samples per wave-tile

    int blocks = (ntiles + 3) / 4;           // ~4 tiles per block (8 waves)
    if (blocks > 2048) blocks = 2048;
    if (blocks < 1) blocks = 1;

    gnn_collapsed_wmma<<<blocks, 256, 0, stream>>>(
        x, W1, b1, W2, b2, W3, b3, W4, b4, linW, linb, out, B, ntiles);
}
